// OTSurv_15857019256984
// MI455X (gfx1250) — matrix operand order repaired and checked by reading, not compiled, not run on hardware
//
#include <hip/hip_runtime.h>

typedef __attribute__((ext_vector_type(16))) __bf16 v16bf;
typedef __attribute__((ext_vector_type(8)))  float  v8f;
typedef __attribute__((ext_vector_type(2)))  float  v2f;
typedef __attribute__((ext_vector_type(4)))  unsigned int v4u;
typedef __attribute__((ext_vector_type(8)))  int    v8i;
typedef __attribute__((ext_vector_type(4)))  int    v4i;

#define NB   8         // batch
#define NN   8192      // patches per slide
#define NP   1024      // input feature dim
#define NH   256       // hidden
#define NK   16        // prototypes
#define NROW (NB*NN)   // 65536 total rows
#define EPSF 0.1f
#define INVE 10.0f
#define SINKIT 50

// ---------------------------------------------------------------------------
// K0: W_enc [1024][256] f32  ->  WencT [256][1024] bf16 (h-major, k contiguous)
// ---------------------------------------------------------------------------
__global__ __launch_bounds__(256) void k_prep(const float* __restrict__ W_enc,
                                              __bf16* __restrict__ WencT) {
  int t = blockIdx.x * 256 + threadIdx.x;          // 0 .. 262143
  if (t < NP * NH) {
    int k = t >> 8;           // 0..1023  (row of W_enc)
    int h = t & 255;          // 0..255   (col of W_enc)
    WencT[(long)h * NP + k] = (__bf16)W_enc[t];
  }
}

// ---------------------------------------------------------------------------
// K0b: zero h-accumulator (2048 f32) and per-batch cost max (8 int-bits)
// ---------------------------------------------------------------------------
__global__ __launch_bounds__(256) void k_init(float* __restrict__ hbuf,
                                              int* __restrict__ maxb) {
  int t = blockIdx.x * 256 + threadIdx.x;
  if (t < NB * NH) hbuf[t] = 0.0f;
  if (t < NB)      maxb[t] = 0;    // bits of +0.0f
}

// ---------------------------------------------------------------------------
// TDM helper: issue tensor_load_to_lds of a 2D tile:
//   tile_dim0 = 16 dwords (64 B), tile_dim1 = 256 rows,
//   row stride = 512 dwords (2048 B), dest LDS contiguous (16 KB).
// ---------------------------------------------------------------------------
__device__ __forceinline__ void tdm_load_wchunk(const __bf16* gsrc, void* ldst) {
#if __has_builtin(__builtin_amdgcn_tensor_load_to_lds)
  unsigned long long ga = (unsigned long long)gsrc;      // 57-bit global addr
  unsigned int lds = (unsigned int)(unsigned long long)ldst; // LDS byte offset
  v4u g0;
  g0[0] = 1u;                                   // count=1, no gather
  g0[1] = lds;                                  // lds_addr [63:32]
  g0[2] = (unsigned int)(ga & 0xffffffffull);   // global_addr [95:64]
  g0[3] = (unsigned int)((ga >> 32) & 0x01ffffffull) | (2u << 30); // [120:96]|type=2
  v8i g1;
  g1[0] = (int)(2u << 16);          // data_size=4B; mask=0; no pad/iterate
  g1[1] = (int)(512u << 16);        // tensor_dim0[15:0]=512 dwords @ [63:48]
  g1[2] = (int)(256u << 16);        // tensor_dim0[31:16]=0, tensor_dim1=256
  g1[3] = (int)(16u << 16);         // tensor_dim1[31:16]=0, tile_dim0=16
  g1[4] = (int)256u;                // tile_dim1=256, tile_dim2=0
  g1[5] = (int)512u;                // tensor_dim0_stride[31:0]=512 dwords
  g1[6] = 0;                        // stride hi / dim1_stride lo
  g1[7] = 0;
  v4i g2 = {0, 0, 0, 0};            // 2D tensor: dims 2/3 unused
  v4i g3 = {0, 0, 0, 0};
#if defined(__clang_major__) && __clang_major__ >= 23
  v8i g4 = {0, 0, 0, 0, 0, 0, 0, 0};
  __builtin_amdgcn_tensor_load_to_lds(g0, g1, g2, g3, g4, 0);
#else
  __builtin_amdgcn_tensor_load_to_lds(g0, g1, g2, g3, 0);
#endif
#else
  (void)gsrc; (void)ldst;
#endif
}

// ---------------------------------------------------------------------------
// K1: enc = relu(features @ W_enc + b_enc)   [65536,1024]x[1024,256]
//     bf16 WMMA, f32 accumulate. Block = 128 thr (4 waves); each wave does
//     16 rows x 256 cols (16 WMMA tiles).  K-loop: 32 steps of K=32.
//     W chunks: TDM tensor_load_to_lds, double-buffered (wave 0 issues the
//     next chunk while all waves compute the current one).
// ---------------------------------------------------------------------------
__global__ __launch_bounds__(128) void k_enc(const float* __restrict__ feat,
                                             const __bf16* __restrict__ WencT,
                                             const float* __restrict__ b_enc,
                                             float* __restrict__ enc) {
  __shared__ __bf16 sB[2][32 * NH];         // two 16 KB chunks: [h][32] bf16
  const int tid  = threadIdx.x;
  const int wave = tid >> 5;
  const int lane = tid & 31;
  const int hi   = lane >> 4;               // half-wave select
  const int lm   = lane & 15;
  const long rowBase = (long)blockIdx.x * 64 + wave * 16;

  v8f acc[16];
  const v8f vz = {0.f,0.f,0.f,0.f,0.f,0.f,0.f,0.f};
#pragma unroll
  for (int t = 0; t < 16; ++t) acc[t] = vz;

  const float* arow = feat + (rowBase + lm) * NP;

#if __has_builtin(__builtin_amdgcn_tensor_load_to_lds)
  // prologue: TDM chunk 0 -> buffer 0
  if (wave == 0) {
    tdm_load_wchunk(WencT, &sB[0][0]);
    __builtin_amdgcn_s_wait_tensorcnt(0);
  }
  __syncthreads();
#else
  // fallback: plain cooperative staging of chunk 0
  {
    for (int r = 0; r < 2; ++r) {
      int h = tid * 2 + r;
      const float4* src = (const float4*)(WencT + (long)h * NP);
      float4* dst = (float4*)(&sB[0][0] + h * 32);
      dst[0] = src[0];
      dst[1] = src[1];
    }
    __syncthreads();
  }
#endif

  for (int ks = 0; ks < 32; ++ks) {
    const int kb = ks * 32;
    const int cur = ks & 1;
    const __bf16* sBc = &sB[cur][0];

#if __has_builtin(__builtin_amdgcn_tensor_load_to_lds)
    if (wave == 0 && ks + 1 < 32)           // kick off next chunk into other buf
      tdm_load_wchunk(WencT + (long)(ks + 1) * 32, &sB[cur ^ 1][0]);
#endif

    // A fragment (16x32 bf16): per-lane k = 16*hi + j
    const float* ap = arow + kb;
    __builtin_prefetch(ap + 64, 0, 0);       // global_prefetch next chunks
    float4 a0 = *(const float4*)(ap + hi * 16);
    float4 a1 = *(const float4*)(ap + hi * 16 + 4);
    float4 a2 = *(const float4*)(ap + hi * 16 + 8);
    float4 a3 = *(const float4*)(ap + hi * 16 + 12);
    v16bf A;
    A[0]=(__bf16)a0.x;  A[1]=(__bf16)a0.y;  A[2]=(__bf16)a0.z;  A[3]=(__bf16)a0.w;
    A[4]=(__bf16)a1.x;  A[5]=(__bf16)a1.y;  A[6]=(__bf16)a1.z;  A[7]=(__bf16)a1.w;
    A[8]=(__bf16)a2.x;  A[9]=(__bf16)a2.y;  A[10]=(__bf16)a2.z; A[11]=(__bf16)a2.w;
    A[12]=(__bf16)a3.x; A[13]=(__bf16)a3.y; A[14]=(__bf16)a3.z; A[15]=(__bf16)a3.w;

    // preload ALL B fragments (one ds clause, one wait), then WMMA chain
    v16bf Bf[16];
#pragma unroll
    for (int t = 0; t < 16; ++t)
      Bf[t] = *(const v16bf*)(sBc + (t * 16 + lm) * 32 + 16 * hi);
#pragma unroll
    for (int t = 0; t < 16; ++t)
      acc[t] = __builtin_amdgcn_wmma_f32_16x16x32_bf16(
          false, A, false, Bf[t], (short)0, acc[t], false, false);

#if __has_builtin(__builtin_amdgcn_tensor_load_to_lds)
    if (wave == 0) __builtin_amdgcn_s_wait_tensorcnt(0);
    __syncthreads();                         // next buffer visible to all waves
#else
    __syncthreads();
    if (ks + 1 < 32) {
      for (int r = 0; r < 2; ++r) {
        int h = tid * 2 + r;
        const float4* src = (const float4*)(WencT + (long)h * NP + kb + 32);
        float4* dst = (float4*)(&sB[cur ^ 1][0] + h * 32);
        dst[0] = src[0];
        dst[1] = src[1];
      }
      __syncthreads();
    }
#endif
  }

  // epilogue: + bias, relu (also maps NaN -> 0 == nan_to_num after relu)
#pragma unroll
  for (int t = 0; t < 16; ++t) {
    const int h = t * 16 + lm;
    const float bb = b_enc[h];
#pragma unroll
    for (int r = 0; r < 8; ++r) {
      long row = rowBase + r + 8 * hi;
      float v = acc[t][r] + bb;
      v = (v > 0.0f) ? v : 0.0f;
      enc[row * NH + h] = v;
    }
  }
}

// ---------------------------------------------------------------------------
// K2: cost C[row][k] = sum_h (enc - proto)^2  (== max(sq,0)); per-batch max
// ---------------------------------------------------------------------------
__global__ __launch_bounds__(256) void k_cost(const float* __restrict__ enc,
                                              const float* __restrict__ protos,
                                              float* __restrict__ C,
                                              int* __restrict__ maxb) {
  __shared__ float se[16][NH];
  __shared__ float sp[NK][NH];
  __shared__ float red[256];
  const int tid = threadIdx.x;
  const long rowBase = (long)blockIdx.x * 16;

  for (int i = tid; i < 16 * NH; i += 256) {
    se[i >> 8][i & 255] = enc[(rowBase + (i >> 8)) * NH + (i & 255)];
    sp[i >> 8][i & 255] = protos[i];
  }
  __syncthreads();

  const int r = tid & 15;       // row within tile
  const int p = tid >> 4;       // prototype
  float c = 0.0f;
#pragma unroll 8
  for (int h = 0; h < NH; ++h) {
    float d = se[r][h] - sp[p][h];
    c = fmaf(d, d, c);
  }
  C[(rowBase + r) * NK + p] = c;

  red[tid] = c;
  __syncthreads();
  for (int s = 128; s > 0; s >>= 1) {
    if (tid < s) red[tid] = fmaxf(red[tid], red[tid + s]);
    __syncthreads();
  }
  if (tid == 0) {
    int b = blockIdx.x >> 9;            // 512 blocks per batch
    atomicMax(maxb + b, __float_as_int(red[0]));   // c >= 0: int-max works
  }
}

// ---------------------------------------------------------------------------
// K3: Sinkhorn. 8 blocks (one per batch) x 1024 threads; each thread owns
//     8 rows. Normalized cost cached in registers; f in registers; g shared
//     via LDS. Writes Tt[b][k][n] transposed for the f32-WMMA agg GEMM.
// ---------------------------------------------------------------------------
__global__ __launch_bounds__(1024) void k_sink(const float* __restrict__ C,
                                               const int* __restrict__ maxb,
                                               float* __restrict__ Tt) {
  const int b = blockIdx.x, tid = threadIdx.x;
  __shared__ float gS[NK];
  __shared__ float redM[32 * NK];
  __shared__ float redS[32 * NK];

  const float invMax = 1.0f / __int_as_float(maxb[b]);
  float Cr[8][NK];
#pragma unroll
  for (int i = 0; i < 8; ++i) {
    long n = tid + (long)i * 1024;
    const float4* p = (const float4*)(C + ((long)b * NN + n) * NK);
    float4 c0 = p[0], c1 = p[1], c2 = p[2], c3 = p[3];
    Cr[i][0]=c0.x*invMax;  Cr[i][1]=c0.y*invMax;  Cr[i][2]=c0.z*invMax;  Cr[i][3]=c0.w*invMax;
    Cr[i][4]=c1.x*invMax;  Cr[i][5]=c1.y*invMax;  Cr[i][6]=c1.z*invMax;  Cr[i][7]=c1.w*invMax;
    Cr[i][8]=c2.x*invMax;  Cr[i][9]=c2.y*invMax;  Cr[i][10]=c2.z*invMax; Cr[i][11]=c2.w*invMax;
    Cr[i][12]=c3.x*invMax; Cr[i][13]=c3.y*invMax; Cr[i][14]=c3.z*invMax; Cr[i][15]=c3.w*invMax;
  }
  float f[8];
#pragma unroll
  for (int i = 0; i < 8; ++i) f[i] = 0.0f;
  if (tid < NK) gS[tid] = 0.0f;
  __syncthreads();

  const float EA = EPSF * (-9.0109131f);   // eps * log(1/N)
  const float EB = EPSF * (-2.7725887f);   // eps * log(1/K)
  const int wv = tid >> 5, ln = tid & 31;

  for (int it = 0; it < SINKIT; ++it) {
    float gl[NK];
#pragma unroll
    for (int k = 0; k < NK; ++k) gl[k] = gS[k];

    // f-update
#pragma unroll
    for (int i = 0; i < 8; ++i) {
      float m = -1e30f;
#pragma unroll
      for (int k = 0; k < NK; ++k) m = fmaxf(m, (gl[k] - Cr[i][k]) * INVE);
      float s = 0.0f;
#pragma unroll
      for (int k = 0; k < NK; ++k) s += __expf((gl[k] - Cr[i][k]) * INVE - m);
      f[i] = EA - EPSF * (m + __logf(s));
    }

    // g-update: online logsumexp over n, per prototype k
    float pm[NK], ps[NK];
#pragma unroll
    for (int k = 0; k < NK; ++k) { pm[k] = -1e30f; ps[k] = 0.0f; }
#pragma unroll
    for (int i = 0; i < 8; ++i) {
#pragma unroll
      for (int k = 0; k < NK; ++k) {
        float t = (f[i] - Cr[i][k]) * INVE;
        float m = fmaxf(pm[k], t);
        ps[k] = ps[k] * __expf(pm[k] - m) + __expf(t - m);
        pm[k] = m;
      }
    }
#pragma unroll
    for (int off = 16; off > 0; off >>= 1) {
#pragma unroll
      for (int k = 0; k < NK; ++k) {
        float om = __shfl_xor(pm[k], off, 32);
        float os = __shfl_xor(ps[k], off, 32);
        float nm = fmaxf(pm[k], om);
        ps[k] = ps[k] * __expf(pm[k] - nm) + os * __expf(om - nm);
        pm[k] = nm;
      }
    }
    __syncthreads();
    if (ln == 0) {
#pragma unroll
      for (int k = 0; k < NK; ++k) { redM[wv * NK + k] = pm[k]; redS[wv * NK + k] = ps[k]; }
    }
    __syncthreads();
    if (tid < NK) {
      float m = -1e30f, s = 0.0f;
      for (int w = 0; w < 32; ++w) {
        float om = redM[w * NK + tid], os = redS[w * NK + tid];
        float nm = fmaxf(m, om);
        s = s * __expf(m - nm) + os * __expf(om - nm);
        m = nm;
      }
      gS[tid] = EB - EPSF * (m + __logf(s));
    }
    __syncthreads();
  }

  float gl[NK];
#pragma unroll
  for (int k = 0; k < NK; ++k) gl[k] = gS[k];
#pragma unroll
  for (int i = 0; i < 8; ++i) {
    long n = tid + (long)i * 1024;
#pragma unroll
    for (int k = 0; k < NK; ++k)
      Tt[((long)b * NK + k) * NN + n] = __expf((f[i] + gl[k] - Cr[i][k]) * INVE);
  }
}

// ---------------------------------------------------------------------------
// K4: agg[b] = T^T @ enc[b], f32 WMMA 16x16x4, fused with W_lin contraction.
// ---------------------------------------------------------------------------
__global__ __launch_bounds__(128) void k_agg(const float* __restrict__ Tt,
                                             const float* __restrict__ enc,
                                             const float* __restrict__ W_lin,
                                             float* __restrict__ hbuf) {
  const int b    = blockIdx.x;
  const int wave = threadIdx.x >> 5;
  const int lane = threadIdx.x & 31;
  const int hi   = lane >> 4;
  const int lm   = lane & 15;
  const int hBase = blockIdx.y * 64 + wave * 16;

  v8f acc = {0.f,0.f,0.f,0.f,0.f,0.f,0.f,0.f};
  const float* Trow = Tt  + ((long)b * NK + lm) * NN;      // A row m = lm
  const float* ebas = enc + (long)b * NN * NH;

  for (int nb = 0; nb < NN; nb += 4) {
    v2f A;
    A.x = Trow[nb + 2 * hi];
    A.y = Trow[nb + 2 * hi + 1];
    v2f Bv;
    Bv.x = ebas[(long)(nb + 2 * hi)     * NH + hBase + lm];
    Bv.y = ebas[(long)(nb + 2 * hi + 1) * NH + hBase + lm];
    __builtin_prefetch(&ebas[(long)(nb + 8) * NH + hBase + lm], 0, 0);
    acc = __builtin_amdgcn_wmma_f32_16x16x4_f32(
        false, A, false, Bv, (short)0, acc, false, false);
  }

  float part = 0.0f;
#pragma unroll
  for (int r = 0; r < 8; ++r) part += acc[r] * W_lin[r + 8 * hi];
  atomicAdd(hbuf + b * NH + hBase + lm, part);   // lane pairs (L, L+16) share h
}

// ---------------------------------------------------------------------------
// K5: logits[b] = sum_h (h[b][h] + b_lin) * W_cls[h] + b_cls
// ---------------------------------------------------------------------------
__global__ __launch_bounds__(256) void k_final(const float* __restrict__ hbuf,
                                               const float* __restrict__ b_lin,
                                               const float* __restrict__ W_cls,
                                               const float* __restrict__ b_cls,
                                               float* __restrict__ out) {
  __shared__ float red[256];
  const int b = blockIdx.x, h = threadIdx.x;
  red[h] = (hbuf[b * NH + h] + b_lin[0]) * W_cls[h];
  __syncthreads();
  for (int s = 128; s > 0; s >>= 1) {
    if (h < s) red[h] += red[h + s];
    __syncthreads();
  }
  if (h == 0) out[b] = red[0] + b_cls[0];
}

// ---------------------------------------------------------------------------
extern "C" void kernel_launch(void* const* d_in, const int* in_sizes, int n_in,
                              void* d_out, int out_size, void* d_ws, size_t ws_size,
                              hipStream_t stream) {
  (void)in_sizes; (void)n_in; (void)out_size; (void)ws_size;
  const float* feat   = (const float*)d_in[0];   // [8,8192,1024]
  const float* W_enc  = (const float*)d_in[1];   // [1024,256]
  const float* b_enc  = (const float*)d_in[2];   // [256]
  const float* protos = (const float*)d_in[3];   // [16,256]
  const float* W_lin  = (const float*)d_in[4];   // [1,16]
  const float* b_lin  = (const float*)d_in[5];   // [1]
  const float* W_cls  = (const float*)d_in[6];   // [1,256]
  const float* b_cls  = (const float*)d_in[7];   // [1]
  float* out = (float*)d_out;                    // [8,1]

  char* ws = (char*)d_ws;
  const size_t OFF_ENC   = 0;                                  // 64 MB
  const size_t OFF_C     = OFF_ENC + (size_t)NROW * NH * 4;    // 4 MB
  const size_t OFF_TT    = OFF_C   + (size_t)NROW * NK * 4;    // 4 MB
  const size_t OFF_WT    = OFF_TT  + (size_t)NB * NK * NN * 4; // 512 KB
  const size_t OFF_MAX   = OFF_WT  + (size_t)NH * NP * 2;      // 32 B (+pad)
  const size_t OFF_HBUF  = OFF_MAX + 256;                      // 8 KB
  float*  enc   = (float*)(ws + OFF_ENC);
  float*  C     = (float*)(ws + OFF_C);
  float*  Tt    = (float*)(ws + OFF_TT);
  __bf16* WencT = (__bf16*)(ws + OFF_WT);
  int*    maxb  = (int*)(ws + OFF_MAX);
  float*  hbuf  = (float*)(ws + OFF_HBUF);

  k_prep <<<dim3((NP * NH + 255) / 256), dim3(256), 0, stream>>>(W_enc, WencT);
  k_init <<<dim3(8), dim3(256), 0, stream>>>(hbuf, maxb);
  k_enc  <<<dim3(NROW / 64), dim3(128), 0, stream>>>(feat, WencT, b_enc, enc);
  k_cost <<<dim3(NROW / 16), dim3(256), 0, stream>>>(enc, protos, C, maxb);
  k_sink <<<dim3(NB), dim3(1024), 0, stream>>>(C, maxb, Tt);
  k_agg  <<<dim3(NB, NH / 64), dim3(128), 0, stream>>>(Tt, enc, W_lin, hbuf);
  k_final<<<dim3(NB), dim3(256), 0, stream>>>(hbuf, b_lin, W_cls, b_cls, out);
}
